// SlotAttention_64622077935823
// MI455X (gfx1250) — compile-verified, matrix operand order checked by reading, not run on hardware
//
#include <hip/hip_runtime.h>
#include <hip/hip_bf16.h>

// ---------------------------------------------------------------------------
// Slot Attention, MI455X (gfx1250, wave32, WMMA bf16 16x16x32)
// B=32, N=4096, D=256, S=7, ITERS=3
// ---------------------------------------------------------------------------

typedef __bf16 bf16;
typedef __attribute__((ext_vector_type(8)))  __bf16 bf16x8;
typedef __attribute__((ext_vector_type(16))) __bf16 bf16x16;
typedef __attribute__((ext_vector_type(8)))  float  f32x8;

constexpr int cB = 32;
constexpr int cN = 4096;
constexpr int cD = 256;
constexpr int cS = 7;
constexpr int ITERS = 3;
constexpr float EPS_LN = 1e-5f;
constexpr int XPITCH = cD + 8;   // LDS row pitch (bf16 elems) to dodge bank conflicts

// Load a 16-half WMMA fragment: halves 0-7 at p, halves 8-15 at p+16 (K layout
// for 16-bit A/B fragments: per lane, K j..j+7 then K j+16..j+23).
__device__ __forceinline__ bf16x16 load_frag(const bf16* p) {
  bf16x8 lo = *(const bf16x8*)(p);
  bf16x8 hi = *(const bf16x8*)(p + 16);
  return __builtin_shufflevector(lo, hi, 0,1,2,3,4,5,6,7,8,9,10,11,12,13,14,15);
}

__device__ __forceinline__ f32x8 wmma_bf16(bf16x16 a, bf16x16 b, f32x8 c) {
  return __builtin_amdgcn_wmma_f32_16x16x32_bf16(false, a, false, b, (short)0, c,
                                                 false, false);
}

// ---------------------------------------------------------------------------
// K0: pack a [D,D] fp32 weight (used as x @ W^T) into bf16 WMMA-B fragments.
// Fragment (kt, nt), lane l: 16 halves contiguous at ((kt*16+nt)*32+l)*16.
// B(k=e, n=dout) = W[dout][e];  n = l&15, kbase = (l<16 ? 0 : 8).
// ---------------------------------------------------------------------------
__global__ void k_pack_w(const float* __restrict__ w, bf16* __restrict__ out) {
  int idx  = blockIdx.x * blockDim.x + threadIdx.x;  // 0 .. 8*16*32-1
  int lane = idx & 31;
  int nt   = (idx >> 5) & 15;
  int kt   = idx >> 9;
  int n    = lane & 15;
  int kb   = (lane < 16) ? 0 : 8;
  int dout = nt * 16 + n;
  bf16* o  = out + (size_t)idx * 16;
#pragma unroll
  for (int h = 0; h < 8; ++h) {
    o[h]     = (bf16)w[(size_t)dout * cD + kt * 32 + kb + h];
    o[h + 8] = (bf16)w[(size_t)dout * cD + kt * 32 + kb + 16 + h];
  }
}

// ---------------------------------------------------------------------------
// K1: slots = mu + softplus(sigma) * noise
// ---------------------------------------------------------------------------
__global__ void k_slots_init(const float* __restrict__ mu, const float* __restrict__ sigma,
                             const float* __restrict__ noise, float* __restrict__ slots) {
  int i = blockIdx.x * 256 + threadIdx.x;
  if (i >= cB * cS * cD) return;
  int d = i & (cD - 1);
  float sg = log1pf(__expf(sigma[d]));
  slots[i] = mu[d] + sg * noise[i];
}

// ---------------------------------------------------------------------------
// K2: fused LayerNorm(inputs) -> bf16 LDS tile -> WMMA K and V projections.
// Grid: (N/32, B). Block 256 = 8 waves. Tile: 32 rows x 256 cols, both K & V.
// Wave w: sel=w>>2 (0:K,1:V), 4 n-tiles, 2 m-tiles -> 8 C fragments.
// K written row-major [B][N][D] bf16; V written transposed [B][D][N] bf16.
// ---------------------------------------------------------------------------
__global__ __launch_bounds__(256) void k_ln_kv(
    const float* __restrict__ inp, const float* __restrict__ g, const float* __restrict__ be,
    const bf16* __restrict__ wkf, const bf16* __restrict__ wvf,
    const float* __restrict__ bk, const float* __restrict__ bv,
    bf16* __restrict__ k_out, bf16* __restrict__ vT_out) {
  __shared__ bf16 xs[32 * XPITCH];
  int b  = blockIdx.y;
  int n0 = blockIdx.x * 32;
  int t  = threadIdx.x;

  // ---- LayerNorm: thread t handles row t>>3, 32-elem segment (t&7) ----
  {
    int row = t >> 3, seg = t & 7;
    const float* xr = inp + (((size_t)b * cN + n0 + row) * cD) + seg * 32;
    float vals[32], s1 = 0.f, s2 = 0.f;
#pragma unroll
    for (int i = 0; i < 32; ++i) { float x = xr[i]; vals[i] = x; s1 += x; s2 += x * x; }
#pragma unroll
    for (int m = 1; m < 8; m <<= 1) { s1 += __shfl_xor(s1, m, 8); s2 += __shfl_xor(s2, m, 8); }
    float mean = s1 * (1.f / cD);
    float var  = s2 * (1.f / cD) - mean * mean;
    float rs   = rsqrtf(var + EPS_LN);
    bf16* xd = xs + row * XPITCH + seg * 32;
#pragma unroll
    for (int i = 0; i < 32; ++i) {
      int c = seg * 32 + i;
      xd[i] = (bf16)(g[c] * (vals[i] - mean) * rs + be[c]);
    }
  }
  __syncthreads();

  // ---- WMMA GEMM: x(32x256) @ W^T(256x256) for one of {K, V} per wave ----
  int w = t >> 5, lane = t & 31;
  int sel = w >> 2;               // 0 = K, 1 = V
  int ntb = (w & 3) * 4;          // first of 4 n-tiles
  const bf16* wf = sel ? wvf : wkf;
  int mrow = lane & 15;
  int kb   = (lane < 16) ? 0 : 8;

  f32x8 zero = {0.f, 0.f, 0.f, 0.f, 0.f, 0.f, 0.f, 0.f};
  f32x8 acc[2][4];
#pragma unroll
  for (int mt = 0; mt < 2; ++mt)
#pragma unroll
    for (int nt = 0; nt < 4; ++nt) acc[mt][nt] = zero;

  for (int kt = 0; kt < cD / 32; ++kt) {
    bf16x16 afr[2];
#pragma unroll
    for (int mt = 0; mt < 2; ++mt)
      afr[mt] = load_frag(xs + (mt * 16 + mrow) * XPITCH + kt * 32 + kb);
#pragma unroll
    for (int nt = 0; nt < 4; ++nt) {
      bf16x16 bfr = *(const bf16x16*)(wf + (((size_t)(kt * 16 + ntb + nt)) * 32 + lane) * 16);
#pragma unroll
      for (int mt = 0; mt < 2; ++mt)
        acc[mt][nt] = wmma_bf16(afr[mt], bfr, acc[mt][nt]);
    }
  }

  const float* bias = sel ? bv : bk;
#pragma unroll
  for (int mt = 0; mt < 2; ++mt)
#pragma unroll
    for (int nt = 0; nt < 4; ++nt) {
      int col = (ntb + nt) * 16 + (lane & 15);
      float bb = bias[col];
#pragma unroll
      for (int i = 0; i < 8; ++i) {
        int m = mt * 16 + i + ((lane < 16) ? 0 : 8);
        float v = acc[mt][nt][i] + bb;
        if (sel == 0)
          k_out[((size_t)b * cN + n0 + m) * cD + col] = (bf16)v;
        else
          vT_out[((size_t)b * cD + col) * (size_t)cN + n0 + m] = (bf16)v;
      }
    }
}

// ---------------------------------------------------------------------------
// K3: q = LN(slots) @ wq^T + bq, written bf16 padded to 16 rows [B][16][D].
// One block per (b, s_pad). Scalar VALU (tiny GEMM: 224x256x256).
// ---------------------------------------------------------------------------
__global__ __launch_bounds__(256) void k_qproj(
    const float* __restrict__ slots, const float* __restrict__ g, const float* __restrict__ be,
    const float* __restrict__ wq, const float* __restrict__ bq, bf16* __restrict__ q_bf) {
  int bs = blockIdx.x, b = bs >> 4, s = bs & 15;
  int t = threadIdx.x;
  if (s >= cS) { q_bf[((size_t)b * 16 + s) * cD + t] = (bf16)0.f; return; }
  __shared__ float xn[cD];
  __shared__ float rs1[8], rs2[8];
  const float* row = slots + ((size_t)b * cS + s) * cD;
  float x = row[t], s1 = x, s2 = x * x;
#pragma unroll
  for (int m = 1; m < 32; m <<= 1) { s1 += __shfl_xor(s1, m, 32); s2 += __shfl_xor(s2, m, 32); }
  if ((t & 31) == 0) { rs1[t >> 5] = s1; rs2[t >> 5] = s2; }
  __syncthreads();
  float S1 = 0.f, S2 = 0.f;
#pragma unroll
  for (int i = 0; i < 8; ++i) { S1 += rs1[i]; S2 += rs2[i]; }
  float mean = S1 * (1.f / cD), var = S2 * (1.f / cD) - mean * mean;
  float rr = rsqrtf(var + EPS_LN);
  xn[t] = g[t] * (x - mean) * rr + be[t];
  __syncthreads();
  float acc = bq[t];
  const float* wr = wq + (size_t)t * cD;
  for (int e = 0; e < cD; ++e) acc += xn[e] * wr[e];
  q_bf[((size_t)b * 16 + s) * cD + t] = (bf16)acc;
}

// ---------------------------------------------------------------------------
// K4: logits[b,s,n] = scale * sum_d q[b,s,d]*k[b,n,d]  (WMMA, M=16 padded)
// Grid (N/128, B); 8 waves, one 16x16 N-tile per wave; K-loop over D.
// k_bf row-major is exactly B-fragment-contiguous here.
// ---------------------------------------------------------------------------
__global__ __launch_bounds__(256) void k_logits(
    const bf16* __restrict__ q_bf, const bf16* __restrict__ k_bf, float* __restrict__ logits) {
  int b = blockIdx.y;
  int w = threadIdx.x >> 5, lane = threadIdx.x & 31;
  int n0 = blockIdx.x * 128 + w * 16;
  int mrow = lane & 15, kb = (lane < 16) ? 0 : 8;
  const bf16* qrow = q_bf + ((size_t)b * 16 + mrow) * cD;
  const bf16* krow = k_bf + ((size_t)b * cN + n0 + mrow) * cD;
  f32x8 acc = {0.f, 0.f, 0.f, 0.f, 0.f, 0.f, 0.f, 0.f};
#pragma unroll
  for (int kt = 0; kt < cD / 32; ++kt) {
    bf16x16 a  = load_frag(qrow + kt * 32 + kb);
    bf16x16 bb = load_frag(krow + kt * 32 + kb);
    acc = wmma_bf16(a, bb, acc);
  }
  const float scale = 0.0625f;  // D^-0.5 = 1/16
  int col = n0 + (lane & 15);
#pragma unroll
  for (int i = 0; i < 8; ++i) {
    int m = i + ((lane < 16) ? 0 : 8);
    if (m < cS) logits[((size_t)b * cS + m) * cN + col] = acc[i] * scale;
  }
}

// ---------------------------------------------------------------------------
// K5: softmax over slots (axis S) per (b,n), in place, + 1e-8.
// ---------------------------------------------------------------------------
__global__ void k_softmax_s(float* __restrict__ attn) {
  int i = blockIdx.x * 256 + threadIdx.x;
  if (i >= cB * cN) return;
  int b = i / cN, n = i - b * cN;
  float* p = attn + (size_t)b * cS * cN + n;
  float v[cS], mx = -1e30f;
#pragma unroll
  for (int s = 0; s < cS; ++s) { v[s] = p[(size_t)s * cN]; mx = fmaxf(mx, v[s]); }
  float sum = 0.f;
#pragma unroll
  for (int s = 0; s < cS; ++s) { v[s] = __expf(v[s] - mx); sum += v[s]; }
  float inv = 1.f / sum;
#pragma unroll
  for (int s = 0; s < cS; ++s) p[(size_t)s * cN] = v[s] * inv + 1e-8f;
}

// ---------------------------------------------------------------------------
// K6: normalize each (b,s) row over N; emit fp32 in place + bf16 padded copy.
// ---------------------------------------------------------------------------
__global__ __launch_bounds__(256) void k_attn_norm(float* __restrict__ attn,
                                                   bf16* __restrict__ attn_bf) {
  int bs = blockIdx.x, b = bs >> 4, s = bs & 15;
  int t = threadIdx.x;
  bf16* ob = attn_bf + ((size_t)b * 16 + s) * cN;
  if (s >= cS) { for (int i = t; i < cN; i += 256) ob[i] = (bf16)0.f; return; }
  float* p = attn + ((size_t)b * cS + s) * cN;
  float sum = 0.f;
  for (int i = t; i < cN; i += 256) sum += p[i];
#pragma unroll
  for (int m = 1; m < 32; m <<= 1) sum += __shfl_xor(sum, m, 32);
  __shared__ float rs[8];
  if ((t & 31) == 0) rs[t >> 5] = sum;
  __syncthreads();
  float tot = 0.f;
#pragma unroll
  for (int i = 0; i < 8; ++i) tot += rs[i];
  float inv = 1.f / tot;
  for (int i = t; i < cN; i += 256) { float a = p[i] * inv; p[i] = a; ob[i] = (bf16)a; }
}

// ---------------------------------------------------------------------------
// K7: updates[b,s,d] = sum_n attn[b,s,n] * v[b,n,d]  (WMMA, K = N = 4096)
// vT layout [B][D][N] makes B fragments contiguous. Grid (D/128, B).
// ---------------------------------------------------------------------------
__global__ __launch_bounds__(256) void k_updates(
    const bf16* __restrict__ attn_bf, const bf16* __restrict__ vT, float* __restrict__ upd) {
  int b = blockIdx.y;
  int w = threadIdx.x >> 5, lane = threadIdx.x & 31;
  int d0 = blockIdx.x * 128 + w * 16;
  int mrow = lane & 15, kb = (lane < 16) ? 0 : 8;
  const bf16* arow = attn_bf + ((size_t)b * 16 + mrow) * cN;
  const bf16* vrow = vT + ((size_t)b * cD + d0 + mrow) * (size_t)cN;
  f32x8 acc = {0.f, 0.f, 0.f, 0.f, 0.f, 0.f, 0.f, 0.f};
  for (int kt = 0; kt < cN / 32; ++kt) {
    const bf16* pa = arow + kt * 32 + kb;
    const bf16* pb = vrow + kt * 32 + kb;
    __builtin_prefetch(pb + 64, 0, 1);   // global_prefetch_b8 two steps ahead
    bf16x16 a  = load_frag(pa);
    bf16x16 bb = load_frag(pb);
    acc = wmma_bf16(a, bb, acc);
  }
  int col = d0 + (lane & 15);
#pragma unroll
  for (int i = 0; i < 8; ++i) {
    int m = i + ((lane < 16) ? 0 : 8);
    if (m < cS) upd[((size_t)b * cS + m) * cD + col] = acc[i];
  }
}

// ---------------------------------------------------------------------------
// K8: GRUCell (torch gate order r,z,n). One block per (b,s) row.
// ---------------------------------------------------------------------------
__global__ __launch_bounds__(256) void k_gru(
    const float* __restrict__ upd, const float* __restrict__ hprev,
    const float* __restrict__ w_ih, const float* __restrict__ b_ih,
    const float* __restrict__ w_hh, const float* __restrict__ b_hh,
    float* __restrict__ hout) {
  int row = blockIdx.x, t = threadIdx.x;
  __shared__ float u[cD], hp[cD];
  u[t]  = upd[(size_t)row * cD + t];
  hp[t] = hprev[(size_t)row * cD + t];
  __syncthreads();
  float ir = b_ih[t], iz = b_ih[t + cD], in_ = b_ih[t + 2 * cD];
  float hr = b_hh[t], hz = b_hh[t + cD], hn  = b_hh[t + 2 * cD];
  const float* wr = w_ih + (size_t)t * cD;
  const float* wz = w_ih + (size_t)(t + cD) * cD;
  const float* wn = w_ih + (size_t)(t + 2 * cD) * cD;
  const float* vr = w_hh + (size_t)t * cD;
  const float* vz = w_hh + (size_t)(t + cD) * cD;
  const float* vn = w_hh + (size_t)(t + 2 * cD) * cD;
  for (int e = 0; e < cD; ++e) {
    float ue = u[e], he = hp[e];
    ir += ue * wr[e]; iz += ue * wz[e]; in_ += ue * wn[e];
    hr += he * vr[e]; hz += he * vz[e]; hn  += he * vn[e];
  }
  float r = 1.f / (1.f + __expf(-(ir + hr)));
  float z = 1.f / (1.f + __expf(-(iz + hz)));
  float nn = tanhf(in_ + r * hn);
  hout[(size_t)row * cD + t] = (1.f - z) * nn + z * hp[t];
}

// ---------------------------------------------------------------------------
// K9: slots += relu(LN(slots) @ w1^T + b1) @ w2^T + b2. One block per row.
// ---------------------------------------------------------------------------
__global__ __launch_bounds__(256) void k_mlp(
    const float* __restrict__ sin_, const float* __restrict__ g, const float* __restrict__ be,
    const float* __restrict__ w1, const float* __restrict__ b1,
    const float* __restrict__ w2, const float* __restrict__ b2,
    float* __restrict__ sout) {
  int row = blockIdx.x, t = threadIdx.x;
  __shared__ float xn[cD], h[2 * cD];
  __shared__ float rs1[8], rs2[8];
  float x = sin_[(size_t)row * cD + t];
  float s1 = x, s2 = x * x;
#pragma unroll
  for (int m = 1; m < 32; m <<= 1) { s1 += __shfl_xor(s1, m, 32); s2 += __shfl_xor(s2, m, 32); }
  if ((t & 31) == 0) { rs1[t >> 5] = s1; rs2[t >> 5] = s2; }
  __syncthreads();
  float S1 = 0.f, S2 = 0.f;
#pragma unroll
  for (int i = 0; i < 8; ++i) { S1 += rs1[i]; S2 += rs2[i]; }
  float mean = S1 * (1.f / cD), var = S2 * (1.f / cD) - mean * mean;
  float rr = rsqrtf(var + EPS_LN);
  xn[t] = g[t] * (x - mean) * rr + be[t];
  __syncthreads();
  for (int j = t; j < 2 * cD; j += 256) {
    float acc = b1[j];
    const float* wr = w1 + (size_t)j * cD;
    for (int e = 0; e < cD; ++e) acc += xn[e] * wr[e];
    h[j] = fmaxf(acc, 0.f);
  }
  __syncthreads();
  float acc = b2[t];
  const float* wr = w2 + (size_t)t * (2 * cD);
  for (int e = 0; e < 2 * cD; ++e) acc += h[e] * wr[e];
  sout[(size_t)row * cD + t] = x + acc;
}

// ---------------------------------------------------------------------------
// K10: pack outputs: [slots (B,S,D)] ++ [attn transposed (B,N,S)]
// ---------------------------------------------------------------------------
__global__ void k_finalize(const float* __restrict__ slots, const float* __restrict__ attn,
                           float* __restrict__ out) {
  int i = blockIdx.x * 256 + threadIdx.x;
  const int n_slots = cB * cS * cD;
  const int total = n_slots + cB * cN * cS;
  if (i >= total) return;
  if (i < n_slots) { out[i] = slots[i]; return; }
  int j = i - n_slots;
  int b = j / (cN * cS);
  int rem = j - b * (cN * cS);
  int n = rem / cS, s = rem - n * cS;
  out[i] = attn[((size_t)b * cS + s) * cN + n];
}

// ---------------------------------------------------------------------------
extern "C" void kernel_launch(void* const* d_in, const int* in_sizes, int n_in,
                              void* d_out, int out_size, void* d_ws, size_t ws_size,
                              hipStream_t stream) {
  (void)in_sizes; (void)n_in; (void)out_size; (void)ws_size;
  const float* inputs     = (const float*)d_in[0];
  const float* noise      = (const float*)d_in[1];
  const float* ln_in_g    = (const float*)d_in[2];
  const float* ln_in_b    = (const float*)d_in[3];
  const float* ln_slots_g = (const float*)d_in[4];
  const float* ln_slots_b = (const float*)d_in[5];
  const float* ln_mlp_g   = (const float*)d_in[6];
  const float* ln_mlp_b   = (const float*)d_in[7];
  const float* slots_mu   = (const float*)d_in[8];
  const float* slots_sig  = (const float*)d_in[9];
  const float* wq = (const float*)d_in[10]; const float* bq = (const float*)d_in[11];
  const float* wk = (const float*)d_in[12]; const float* bk = (const float*)d_in[13];
  const float* wv = (const float*)d_in[14]; const float* bv = (const float*)d_in[15];
  const float* w_ih = (const float*)d_in[16]; const float* b_ih = (const float*)d_in[17];
  const float* w_hh = (const float*)d_in[18]; const float* b_hh = (const float*)d_in[19];
  const float* w1 = (const float*)d_in[20]; const float* b1 = (const float*)d_in[21];
  const float* w2 = (const float*)d_in[22]; const float* b2 = (const float*)d_in[23];

  // ---- workspace carve-up (256B aligned) ----
  char* base = (char*)d_ws;
  size_t off = 0;
  auto carve = [&](size_t bytes) -> char* {
    char* p = base + off;
    off = (off + bytes + 255) & ~(size_t)255;
    return p;
  };
  bf16*  k_bf    = (bf16*)carve((size_t)cB * cN * cD * 2);   // 64 MB
  bf16*  vT_bf   = (bf16*)carve((size_t)cB * cD * cN * 2);   // 64 MB
  bf16*  wk_frag = (bf16*)carve((size_t)cD * cD * 2);
  bf16*  wv_frag = (bf16*)carve((size_t)cD * cD * 2);
  float* slotsA  = (float*)carve((size_t)cB * cS * cD * 4);
  float* slotsB  = (float*)carve((size_t)cB * cS * cD * 4);
  bf16*  q_bf    = (bf16*)carve((size_t)cB * 16 * cD * 2);
  float* attn    = (float*)carve((size_t)cB * cS * cN * 4);  // logits -> attn in place
  bf16*  attn_bf = (bf16*)carve((size_t)cB * 16 * cN * 2);
  float* updates = (float*)carve((size_t)cB * cS * cD * 4);

  // ---- one-time prep ----
  k_pack_w<<<16, 256, 0, stream>>>(wk, wk_frag);
  k_pack_w<<<16, 256, 0, stream>>>(wv, wv_frag);
  k_slots_init<<<(cB * cS * cD + 255) / 256, 256, 0, stream>>>(slots_mu, slots_sig, noise, slotsA);
  k_ln_kv<<<dim3(cN / 32, cB), 256, 0, stream>>>(inputs, ln_in_g, ln_in_b,
                                                 wk_frag, wv_frag, bk, bv, k_bf, vT_bf);

  // ---- iterations ----
  for (int it = 0; it < ITERS; ++it) {
    k_qproj<<<cB * 16, 256, 0, stream>>>(slotsA, ln_slots_g, ln_slots_b, wq, bq, q_bf);
    k_logits<<<dim3(cN / 128, cB), 256, 0, stream>>>(q_bf, k_bf, attn);
    k_softmax_s<<<(cB * cN + 255) / 256, 256, 0, stream>>>(attn);
    k_attn_norm<<<cB * 16, 256, 0, stream>>>(attn, attn_bf);
    k_updates<<<dim3(cD / 128, cB), 256, 0, stream>>>(attn_bf, vT_bf, updates);
    k_gru<<<cB * cS, 256, 0, stream>>>(updates, slotsA, w_ih, b_ih, w_hh, b_hh, slotsB);
    k_mlp<<<cB * cS, 256, 0, stream>>>(slotsB, ln_mlp_g, ln_mlp_b, w1, b1, w2, b2, slotsA);
  }

  const int total = cB * cS * cD + cB * cN * cS;
  k_finalize<<<(total + 255) / 256, 256, 0, stream>>>(slotsA, attn, (float*)d_out);
}
